// NeuralMemoryNetwork_74792560492649
// MI455X (gfx1250) — matrix operand order repaired and checked by reading, not compiled, not run on hardware
//
#include <hip/hip_runtime.h>
#include <math.h>

#define E 64
#define M_MATS 512
#define NH 8
#define DH 8
#define BATCH 32768
#define EPS 1e-5f

typedef __attribute__((ext_vector_type(16))) __bf16 v16bf;
typedef __attribute__((ext_vector_type(8)))  float  v8f;
typedef __attribute__((ext_vector_type(8)))  __bf16 bf16x8;
typedef __attribute__((ext_vector_type(4)))  unsigned u32x4;
typedef __attribute__((ext_vector_type(4)))  float    f32x4;

union FragBF { v16bf v; u32x4 q[2]; };
union Pack8  { bf16x8 b; u32x4 q; };

// ---------------- workspace layout (bytes) ----------------
static constexpr size_t OFF_WBF  = 0;                      // mem_w as bf16: 512*64*64*2 = 4 MiB
static constexpr size_t OFF_RM   = 4194304;                // row_mean / avg_w: 512*64 f32
static constexpr size_t OFF_WQE  = OFF_RM  + 131072;       // fused Wq (64x64 f32)
static constexpr size_t OFF_WKE  = OFF_WQE + 16384;
static constexpr size_t OFF_WVE  = OFF_WKE + 16384;
static constexpr size_t OFF_BQE  = OFF_WVE + 16384;        // fused biases (64 f32 each)
static constexpr size_t OFF_BKE  = OFF_BQE + 256;
static constexpr size_t OFF_BVE  = OFF_BKE + 256;
static constexpr size_t OFF_QH   = OFF_BVE + 256;          // qh: 512*64 f32
static constexpr size_t OFF_KHT  = OFF_QH  + 131072;       // kh transposed (H,B,8) f32
static constexpr size_t OFF_VHT  = OFF_KHT + 8388608;      // vh transposed (H,B,8) f32
static constexpr size_t OFF_O    = OFF_VHT + 8388608;      // attention output o: 512*64 f32
static constexpr size_t OFF_AOUT = OFF_O   + 131072;       // attn_out: 512*64 f32

// Async global->LDS 16B copy (CDNA5 GLOBAL_LOAD_ASYNC_TO_LDS_B128, ASYNCcnt).
// GVS addressing: mem = saddr(64) + vaddr(i32). vdst = LDS byte offset.
__device__ __forceinline__ void async_load_b128(unsigned lds_off, unsigned goff,
                                                unsigned long long base)
{
    asm volatile("global_load_async_to_lds_b128 %0, %1, %2"
                 :: "v"(lds_off), "v"(goff), "s"(base)
                 : "memory");
}

// ===========================================================
// 1) Convert mem_w -> bf16, compute row_mean (== avg_w)
// ===========================================================
__global__ __launch_bounds__(256)
void prep_kernel(const float* __restrict__ mem_w, __bf16* __restrict__ wbf,
                 float* __restrict__ rm)
{
    const int m = blockIdx.x;                 // 0..511
    const int t = threadIdx.x;                // 0..255
    const float* src = mem_w + (size_t)m * 4096;

    {   // convert 16 contiguous f32 -> bf16
        const int base = t * 16;
        Pack8 p0, p1;
#pragma unroll
        for (int c = 0; c < 2; ++c) {
            f32x4 a = ((const f32x4*)(src + base))[2 * c];
            f32x4 b = ((const f32x4*)(src + base))[2 * c + 1];
            Pack8& p = c == 0 ? p0 : p1;
#pragma unroll
            for (int j = 0; j < 4; ++j) { p.b[j] = (__bf16)a[j]; p.b[4 + j] = (__bf16)b[j]; }
        }
        u32x4* dst = (u32x4*)(wbf + (size_t)m * 4096 + base);
        dst[0] = p0.q; dst[1] = p1.q;
    }
    if (t < 64) {   // row mean over last axis
        const float* r = src + t * 64;
        float s = 0.f;
#pragma unroll
        for (int c = 0; c < 16; ++c) {
            f32x4 v = ((const f32x4*)r)[c];
            s += v[0] + v[1] + v[2] + v[3];
        }
        rm[m * 64 + t] = s * (1.f / 64.f);
    }
}

// ===========================================================
// 2) Fuse projection pairs: W*_eff = W*_inner @ W*, b*_eff
// ===========================================================
__global__ __launch_bounds__(256)
void weights_kernel(const float* __restrict__ Wq, const float* __restrict__ bq,
                    const float* __restrict__ Wk, const float* __restrict__ bk,
                    const float* __restrict__ Wv, const float* __restrict__ bv,
                    const float* __restrict__ ipw, const float* __restrict__ ipb,
                    float* __restrict__ WqE, float* __restrict__ WkE, float* __restrict__ WvE,
                    float* __restrict__ bqE, float* __restrict__ bkE, float* __restrict__ bvE)
{
    const int t = threadIdx.x;
    const float* Wqi = ipw;
    const float* Wki = ipw + 4096;
    const float* Wvi = ipw + 8192;
#pragma unroll 4
    for (int i = 0; i < 16; ++i) {
        int o = t * 16 + i;
        int r = o >> 6, c = o & 63;
        float sq = 0.f, sk = 0.f, sv = 0.f;
        for (int e = 0; e < 64; ++e) {
            sq += Wqi[r * 64 + e] * Wq[e * 64 + c];
            sk += Wki[r * 64 + e] * Wk[e * 64 + c];
            sv += Wvi[r * 64 + e] * Wv[e * 64 + c];
        }
        WqE[o] = sq; WkE[o] = sk; WvE[o] = sv;
    }
    if (t < 64) {
        float sq = 0.f, sk = 0.f, sv = 0.f;
        for (int e = 0; e < 64; ++e) {
            sq += Wqi[t * 64 + e] * bq[e];
            sk += Wki[t * 64 + e] * bk[e];
            sv += Wvi[t * 64 + e] * bv[e];
        }
        bqE[t] = sq + ipb[t];
        bkE[t] = sk + ipb[64 + t];
        bvE[t] = sv + ipb[128 + t];
    }
}

// ===========================================================
// 3) The scan: 512 chained steps of relu(h @ W^T) + LayerNorm
//    bf16 WMMA (v_wmma_f32_16x16x32_bf16), fp32 accum & LN.
//    W_m staged one step ahead via async global->LDS copies
//    (double-buffered, s_wait_asynccnt), so no global latency
//    sits on the 512-step serial chain.
// ===========================================================
#define HSTR 72   // bf16 row stride for sH/sW (conflict-free fragment loads)
#define SSTR 68   // f32 row stride for LN scratch

__global__ __launch_bounds__(256)
void scan_kernel(const float* __restrict__ x, const __bf16* __restrict__ wbf,
                 const float* __restrict__ ln_g, const float* __restrict__ ln_b,
                 float* __restrict__ mem_out)
{
    __shared__ __bf16 sWb[2][64 * HSTR];      // 2 x 9.0 KB, async-filled
    __shared__ __bf16 sH[128 * HSTR];         // 18.0 KB
    __shared__ float  sGB[2][128];            // LN gamma(0..63)/beta(64..127), double-buffered
    __shared__ float  sS[8 * 16 * SSTR];      // 34.0 KB per-wave LN scratch

    const int t    = threadIdx.x;
    const int wave = t >> 5;
    const int lane = t & 31;
    const int lm   = lane & 15;               // row / column index within half-wave
    const int hf   = lane >> 4;               // half-wave select
    const int wrb  = wave * 16;               // wave's row base in sH
    const int grb  = blockIdx.x * 128;        // block's global row base

    // per-thread async-copy geometry: 2 x 16B chunks of the 8 KB W tile
    const int c0 = 2 * t;
    const unsigned ldsW0 = (unsigned)(uintptr_t)(&sWb[0][0]);
    const unsigned ldsW1 = (unsigned)(uintptr_t)(&sWb[1][0]);
    const unsigned ldsc0 = (unsigned)(((c0)     >> 3) * (HSTR * 2) + ((c0)     & 7) * 16);
    const unsigned ldsc1 = (unsigned)(((c0 + 1) >> 3) * (HSTR * 2) + ((c0 + 1) & 7) * 16);
    const unsigned long long wbase = (unsigned long long)wbf;

    // ---- init h tile from x (bf16) ----
    {
        const float* xr = x + (size_t)(grb + wrb + lm) * E + hf * 32;
        __bf16* hr = sH + (wrb + lm) * HSTR + hf * 32;
#pragma unroll
        for (int ch = 0; ch < 4; ++ch) {
            f32x4 a = ((const f32x4*)xr)[2 * ch];
            f32x4 b = ((const f32x4*)xr)[2 * ch + 1];
            Pack8 p;
#pragma unroll
            for (int j = 0; j < 4; ++j) { p.b[j] = (__bf16)a[j]; p.b[4 + j] = (__bf16)b[j]; }
            ((u32x4*)hr)[ch] = p.q;
        }
    }

    // ---- prologue: async-prefetch W_0 into buf0; LN params for m=0 ----
    {
        const unsigned g0 = (unsigned)(c0 * 16);
        async_load_b128(ldsW0 + ldsc0, g0,      wbase);
        async_load_b128(ldsW0 + ldsc1, g0 + 16, wbase);
    }
    float lnCur = 0.f;
    if (t < 128) lnCur = (t < 64) ? ln_g[t] : ln_b[t - 64];

    for (int m = 0; m < M_MATS; ++m) {
        const int cb = m & 1;
        __syncthreads();                       // B1: step m-1 fully consumed
        if (t < 128) sGB[cb][t] = lnCur;       // publish step-m LN params

        {   // async-prefetch W_{m+1} into the other buffer
            const int pf = (m + 1) & (M_MATS - 1);
            const unsigned base = ((m + 1) & 1) ? ldsW1 : ldsW0;
            const unsigned g0 = (unsigned)(pf * 8192 + c0 * 16);
            async_load_b128(base + ldsc0, g0,      wbase);
            async_load_b128(base + ldsc1, g0 + 16, wbase);
            if (t < 128)
                lnCur = (t < 64) ? ln_g[pf * 64 + t] : ln_b[pf * 64 + (t - 64)];
        }
        // 2 older async copies (W_m) must land; the 2 for W_{m+1} stay in flight
        asm volatile("s_wait_asynccnt 2" ::: "memory");
        __syncthreads();                       // B2: buf[m] + LN params visible to all

        const __bf16* sW  = &sWb[cb][0];
        const float*  sG  = &sGB[cb][0];
        const float*  sLB = &sGB[cb][64];

        // ---- GEMM: out[16x64] = h[16x64] @ W^T, via 4 N-tiles x 2 K-halves ----
        v8f C[4];
        const v8f vz = {0.f, 0.f, 0.f, 0.f, 0.f, 0.f, 0.f, 0.f};
#pragma unroll
        for (int nt = 0; nt < 4; ++nt) C[nt] = vz;

#pragma unroll
        for (int kh = 0; kh < 2; ++kh) {
            // A fragment: M=lm, lane holds 16 consecutive k (k = 16*hf + s)
            FragBF A;
            const __bf16* hrow = sH + (wrb + lm) * HSTR + kh * 32 + hf * 16;
            A.q[0] = *(const u32x4*)hrow;
            A.q[1] = *(const u32x4*)(hrow + 8);
#pragma unroll
            for (int nt = 0; nt < 4; ++nt) {
                // B fragment: B[k][n] = W[n][k]; lane holds 16 consecutive k of W row n
                FragBF Bf;
                const __bf16* wrow = sW + (nt * 16 + lm) * HSTR + kh * 32 + hf * 16;
                Bf.q[0] = ((const u32x4*)wrow)[0];
                Bf.q[1] = ((const u32x4*)wrow)[1];
                C[nt] = __builtin_amdgcn_wmma_f32_16x16x32_bf16(
                    false, A.v, false, Bf.v, (short)0, C[nt], false, false);
            }
        }

        // ---- relu + spill fp32 tile to per-wave scratch ----
        float* sc = sS + wave * (16 * SSTR);
#pragma unroll
        for (int nt = 0; nt < 4; ++nt)
#pragma unroll
            for (int v = 0; v < 8; ++v) {
                float val = C[nt][v];
                val = val > 0.f ? val : 0.f;
                sc[(v + 8 * hf) * SSTR + nt * 16 + lm] = val;
            }
        asm volatile("s_wait_dscnt 0" ::: "memory");   // wave-internal LDS ordering

        // ---- LayerNorm stats: lane sums its half-row, combine across halves ----
        float s = 0.f, s2 = 0.f;
        const float* rowp = sc + lm * SSTR + hf * 32;
#pragma unroll
        for (int ch = 0; ch < 8; ++ch) {
            f32x4 v = ((const f32x4*)rowp)[ch];
            s  += v[0] + v[1] + v[2] + v[3];
            s2 += v[0] * v[0] + v[1] * v[1] + v[2] * v[2] + v[3] * v[3];
        }
        s  += __shfl_xor(s, 16);
        s2 += __shfl_xor(s2, 16);
        const float mu   = s * (1.f / 64.f);
        const float var  = s2 * (1.f / 64.f) - mu * mu;
        const float rstd = rsqrtf(var + EPS);

        // ---- normalize, scale/shift, write back h as bf16 ----
        __bf16* hdst = sH + (wrb + lm) * HSTR + hf * 32;
#pragma unroll
        for (int ch = 0; ch < 4; ++ch) {
            const int c = hf * 32 + ch * 8;
            f32x4 a  = ((const f32x4*)(sc + lm * SSTR + c))[0];
            f32x4 b  = ((const f32x4*)(sc + lm * SSTR + c))[1];
            f32x4 ga = ((const f32x4*)(sG + c))[0];
            f32x4 gb = ((const f32x4*)(sG + c))[1];
            f32x4 ba = ((const f32x4*)(sLB + c))[0];
            f32x4 bb = ((const f32x4*)(sLB + c))[1];
            Pack8 p;
#pragma unroll
            for (int j = 0; j < 4; ++j) {
                p.b[j]     = (__bf16)(((a[j] - mu) * rstd) * ga[j] + ba[j]);
                p.b[4 + j] = (__bf16)(((b[j] - mu) * rstd) * gb[j] + bb[j]);
            }
            ((u32x4*)hdst)[ch] = p.q;
        }
    }

    // ---- write mem_out (fp32) ----
    asm volatile("s_wait_dscnt 0" ::: "memory");
    {
        const __bf16* hr = sH + (wrb + lm) * HSTR + hf * 32;
        float* orow = mem_out + (size_t)(grb + wrb + lm) * E + hf * 32;
#pragma unroll
        for (int ch = 0; ch < 4; ++ch) {
            Pack8 p; p.q = ((const u32x4*)hr)[ch];
            f32x4 o0, o1;
#pragma unroll
            for (int j = 0; j < 4; ++j) { o0[j] = (float)p.b[j]; o1[j] = (float)p.b[4 + j]; }
            ((f32x4*)orow)[2 * ch]     = o0;
            ((f32x4*)orow)[2 * ch + 1] = o1;
        }
    }
}

// ===========================================================
// 4) qh = avg_w @ WqE^T + bqE            (512x64)
// ===========================================================
__global__ __launch_bounds__(256)
void qh_kernel(const float* __restrict__ rm, const float* __restrict__ WqE,
               const float* __restrict__ bqE, float* __restrict__ qh)
{
    const int o = blockIdx.x * 256 + threadIdx.x;      // < 512*64
    const int mm = o >> 6, j = o & 63;
    const float* a = rm + mm * 64;
    const float* w = WqE + j * 64;
    float s = bqE[j];
    for (int e = 0; e < 64; ++e) s += a[e] * w[e];
    qh[o] = s;
}

// ===========================================================
// 5) kh/vh = mem_out @ W*E^T + b*E, stored (H, B, 8)
// ===========================================================
__global__ __launch_bounds__(256)
void kv_kernel(const float* __restrict__ mo,
               const float* __restrict__ WkE, const float* __restrict__ bkE,
               const float* __restrict__ WvE, const float* __restrict__ bvE,
               float* __restrict__ kht, float* __restrict__ vht)
{
    const int o = blockIdx.x * 256 + threadIdx.x;      // < B*64
    const int n = o >> 6, j = o & 63;
    const float* a  = mo  + (size_t)n * 64;
    const float* wk = WkE + j * 64;
    const float* wv = WvE + j * 64;
    float sk = bkE[j], sv = bvE[j];
    for (int e = 0; e < 64; ++e) { sk += a[e] * wk[e]; sv += a[e] * wv[e]; }
    const size_t dst = ((size_t)(j >> 3) * BATCH + n) * 8 + (j & 7);
    kht[dst] = sk;
    vht[dst] = sv;
}

// ===========================================================
// 6) Flash-style attention: one WG per (head, 16 q-rows)
// ===========================================================
__global__ __launch_bounds__(256)
void attn_kernel(const float* __restrict__ qh, const float* __restrict__ kht,
                 const float* __restrict__ vht, float* __restrict__ o_out)
{
    __shared__ float red[16][16][10];
    const int h  = blockIdx.x >> 5;
    const int mt = blockIdx.x & 31;
    const int t  = threadIdx.x;
    const int r  = t >> 4, g = t & 15;
    const int m  = mt * 16 + r;
    const float scale = 0.35355339059327373f;   // 1/sqrt(8)

    float q[8];
#pragma unroll
    for (int d = 0; d < 8; ++d) q[d] = qh[m * 64 + h * 8 + d];

    float mx = -1e30f, sum = 0.f;
    float acc[8] = {0.f, 0.f, 0.f, 0.f, 0.f, 0.f, 0.f, 0.f};
    const float* kb = kht + (size_t)h * BATCH * 8;
    const float* vb = vht + (size_t)h * BATCH * 8;

    for (int n = g; n < BATCH; n += 16) {
        const f32x4* kp = (const f32x4*)(kb + (size_t)n * 8);
        f32x4 k0 = kp[0], k1 = kp[1];
        float s = q[0]*k0[0] + q[1]*k0[1] + q[2]*k0[2] + q[3]*k0[3]
                + q[4]*k1[0] + q[5]*k1[1] + q[6]*k1[2] + q[7]*k1[3];
        s *= scale;
        float nmx  = fmaxf(mx, s);
        float corr = __expf(mx - nmx);
        float p    = __expf(s - nmx);
        const f32x4* vp = (const f32x4*)(vb + (size_t)n * 8);
        f32x4 v0 = vp[0], v1 = vp[1];
        sum = sum * corr + p;
#pragma unroll
        for (int d = 0; d < 4; ++d) {
            acc[d]     = acc[d]     * corr + p * v0[d];
            acc[4 + d] = acc[4 + d] * corr + p * v1[d];
        }
        mx = nmx;
    }

    red[r][g][0] = mx;
    red[r][g][1] = sum;
#pragma unroll
    for (int d = 0; d < 8; ++d) red[r][g][2 + d] = acc[d];
    __syncthreads();

    if (t < 16) {
        const int rr = t;
        float gmx = -1e30f;
        for (int i = 0; i < 16; ++i) gmx = fmaxf(gmx, red[rr][i][0]);
        float gsum = 0.f, ga[8] = {0.f, 0.f, 0.f, 0.f, 0.f, 0.f, 0.f, 0.f};
        for (int i = 0; i < 16; ++i) {
            float w = __expf(red[rr][i][0] - gmx);
            gsum += red[rr][i][1] * w;
#pragma unroll
            for (int d = 0; d < 8; ++d) ga[d] += red[rr][i][2 + d] * w;
        }
        const float inv = 1.f / gsum;
        const int mm = mt * 16 + rr;
#pragma unroll
        for (int d = 0; d < 8; ++d) o_out[mm * 64 + h * 8 + d] = ga[d] * inv;
    }
}

// ===========================================================
// 7) attn_out = o @ out_w^T + out_b       (512x64)
// ===========================================================
__global__ __launch_bounds__(256)
void attnout_kernel(const float* __restrict__ o, const float* __restrict__ out_w,
                    const float* __restrict__ out_b, float* __restrict__ aout)
{
    const int idx = blockIdx.x * 256 + threadIdx.x;    // < 512*64
    const int mm = idx >> 6, j = idx & 63;
    const float* a = o + mm * 64;
    const float* w = out_w + j * 64;
    float s = out_b[j];
    for (int e = 0; e < 64; ++e) s += a[e] * w[e];
    aout[idx] = s;
}

// ===========================================================
// 8) new_mem_w = mem_w * (1 + attn_out / row_mean)
// ===========================================================
__global__ __launch_bounds__(256)
void finalize_kernel(const float* __restrict__ mem_w, const float* __restrict__ aout,
                     const float* __restrict__ rm, float* __restrict__ out)
{
    const int gid = blockIdx.x * 256 + threadIdx.x;    // < 524288
    const size_t base = (size_t)gid * 4;
    const int me = (int)(base >> 6);                   // (m*64 + e)
    const float f = 1.f + aout[me] / rm[me];
    f32x4 w = ((const f32x4*)(mem_w + base))[0];
    f32x4 res;
#pragma unroll
    for (int j = 0; j < 4; ++j) res[j] = w[j] * f;
    ((f32x4*)(out + base))[0] = res;
}

// ===========================================================
extern "C" void kernel_launch(void* const* d_in, const int* in_sizes, int n_in,
                              void* d_out, int out_size, void* d_ws, size_t ws_size,
                              hipStream_t stream)
{
    const float* x    = (const float*)d_in[0];
    const float* memw = (const float*)d_in[1];
    const float* lng  = (const float*)d_in[2];
    const float* lnb  = (const float*)d_in[3];
    const float* Wq   = (const float*)d_in[4];
    const float* bq   = (const float*)d_in[5];
    const float* Wk   = (const float*)d_in[6];
    const float* bk   = (const float*)d_in[7];
    const float* Wv   = (const float*)d_in[8];
    const float* bv   = (const float*)d_in[9];
    const float* ipw  = (const float*)d_in[10];
    const float* ipb  = (const float*)d_in[11];
    const float* outw = (const float*)d_in[12];
    const float* outb = (const float*)d_in[13];

    float* out      = (float*)d_out;
    float* mem_out  = out;                             // [B*E]
    float* new_w    = out + (size_t)BATCH * E;         // [M*E*E]

    char* ws = (char*)d_ws;                            // needs ~21.6 MB
    __bf16* wbf = (__bf16*)(ws + OFF_WBF);
    float* rm   = (float*)(ws + OFF_RM);
    float* WqE  = (float*)(ws + OFF_WQE);
    float* WkE  = (float*)(ws + OFF_WKE);
    float* WvE  = (float*)(ws + OFF_WVE);
    float* bqE  = (float*)(ws + OFF_BQE);
    float* bkE  = (float*)(ws + OFF_BKE);
    float* bvE  = (float*)(ws + OFF_BVE);
    float* qh   = (float*)(ws + OFF_QH);
    float* kht  = (float*)(ws + OFF_KHT);
    float* vht  = (float*)(ws + OFF_VHT);
    float* oo   = (float*)(ws + OFF_O);
    float* aout = (float*)(ws + OFF_AOUT);

    prep_kernel    <<<512,  256, 0, stream>>>(memw, wbf, rm);
    weights_kernel <<<1,    256, 0, stream>>>(Wq, bq, Wk, bk, Wv, bv, ipw, ipb,
                                              WqE, WkE, WvE, bqE, bkE, bvE);
    scan_kernel    <<<256,  256, 0, stream>>>(x, wbf, lng, lnb, mem_out);
    qh_kernel      <<<128,  256, 0, stream>>>(rm, WqE, bqE, qh);
    kv_kernel      <<<8192, 256, 0, stream>>>(mem_out, WkE, bkE, WvE, bvE, kht, vht);
    attn_kernel    <<<256,  256, 0, stream>>>(qh, kht, vht, oo);
    attnout_kernel <<<128,  256, 0, stream>>>(oo, outw, outb, aout);
    finalize_kernel<<<2048, 256, 0, stream>>>(memw, aout, rm, new_w);
}
